// AddTapeToken_23811298689463
// MI455X (gfx1250) — matrix operand order repaired and checked by reading, not compiled, not run on hardware
//
#include <hip/hip_runtime.h>
#include <hip/hip_bf16.h>
#include <stdint.h>

// ---------------- problem constants (match reference) ----------------
#define BS     64
#define DD     768
#define BANK   65536
#define KTOP   16
#define STEPS  16
#define COLS   256                 // score columns per block
#define NBLK   (BANK / COLS)       // 256 blocks in scores kernel
#define MWORDS (BANK / 32)         // 2048 mask words per row
#define INV_SQRT_D 0.03608439182435161f   // 1/sqrt(768)
#define NEG_SENTINEL (-3.0e38f)

typedef __attribute__((ext_vector_type(16))) __bf16 v16bf;
typedef __attribute__((ext_vector_type(8)))  __bf16 v8bf;
typedef __attribute__((ext_vector_type(4)))  __bf16 v4bf;
typedef __attribute__((ext_vector_type(8)))  float  v8f;
typedef __attribute__((ext_vector_type(4)))  float  v4f;
typedef __attribute__((ext_vector_type(4)))  uint32_t v4u;

static __device__ __forceinline__ __bf16 f2bf(float f) {
  union { float f; uint32_t u; } v; v.f = f;
  uint32_t r = v.u + 0x7FFFu + ((v.u >> 16) & 1u);   // round-to-nearest-even
  uint16_t h = (uint16_t)(r >> 16);
  return __builtin_bit_cast(__bf16, h);
}

// ---------------- init: zero state, zero toks, copy q (f32 + bf16) --------
__global__ void init_kernel(const float* __restrict__ q_in,
                            float* __restrict__ q, __bf16* __restrict__ qb,
                            float* __restrict__ hp, float* __restrict__ rem,
                            float* __restrict__ nup, int* __restrict__ flag,
                            uint32_t* __restrict__ maskbits,
                            float* __restrict__ out_toks) {
  const int stride = gridDim.x * blockDim.x;
  for (int i = blockIdx.x * blockDim.x + threadIdx.x;
       i < BS * STEPS * DD; i += stride) {           // 786432 dominates
    out_toks[i] = 0.0f;                              // pre-zero toks output
    if (i < BS * MWORDS) maskbits[i] = 0u;           // 512 KB mask bitset
    if (i < BS * DD) { float v = q_in[i]; q[i] = v; qb[i] = f2bf(v); }
    if (i < BS) { hp[i] = 0.0f; rem[i] = 0.0f; nup[i] = 0.0f; }
    if (i == 0) *flag = 0;
  }
}

// ---------------- one-time f32 -> bf16 bank conversion ----------------
__global__ void convert_kernel(const float* __restrict__ tape,
                               __bf16* __restrict__ tapeb) {
  const int stride = gridDim.x * blockDim.x;
  const int n4 = (BANK * DD) / 4;
  for (int i = blockIdx.x * blockDim.x + threadIdx.x; i < n4; i += stride) {
    v4f f = *(const v4f*)(tape + (size_t)i * 4);
    v4bf b; b[0] = f2bf(f[0]); b[1] = f2bf(f[1]); b[2] = f2bf(f[2]); b[3] = f2bf(f[3]);
    *(v4bf*)(tapeb + (size_t)i * 4) = b;
  }
}

// ---------------- scores GEMM (WMMA bf16) + block-local top-16 --------
template <bool PRE>
__global__ __launch_bounds__(256) void scores_topk_kernel(
    const __bf16* __restrict__ qb, const __bf16* __restrict__ tapeb,
    const float* __restrict__ tapef, const uint32_t* __restrict__ maskbits,
    const int* __restrict__ flag,
    float* __restrict__ cval, int* __restrict__ cidx) {
  if (*flag) return;                                  // skip_step: all halted

  __shared__ float s_sc[BS][COLS];                    // 64 KB score tile

  const int tid  = threadIdx.x;
  const int lane = tid & 31;
  const int wv   = tid >> 5;                          // wave 0..7
  const int l16  = lane & 15;
  const int hsel = lane >> 4;                         // 0 | 1 (lane half)
  const int cb   = blockIdx.x * COLS;                 // global column base
  const int n0   = cb + wv * 32;                      // this wave's 32 cols

  v8f acc[4][2];
  for (int m = 0; m < 4; ++m)
    for (int t = 0; t < 2; ++t)
      acc[m][t] = (v8f){};

  for (int kb = 0; kb < DD; kb += 32) {
    // A fragments: 16x32 bf16; lane<16 holds K{0..15}, lane>=16 K{16..31}
    v16bf a[4];
    const int alo = kb + hsel * 16;
    for (int m = 0; m < 4; ++m) {
      const __bf16* ap = qb + (size_t)(m * 16 + l16) * DD + alo;
      v8bf lo = *(const v8bf*)ap;
      v8bf hi = *(const v8bf*)(ap + 8);
      v16bf av;
      for (int i = 0; i < 8; ++i) { av[i] = lo[i]; av[8 + i] = hi[i]; }
      a[m] = av;
    }
    // B fragments: 32x16, lane n holds tape row (n0+16t+n), K contiguous by half
    v16bf b[2];
    const int koff = kb + (hsel << 4);
    for (int t = 0; t < 2; ++t) {
      const size_t col = (size_t)(n0 + t * 16 + l16);
      if (PRE) {
        b[t] = *(const v16bf*)(tapeb + col * DD + koff);
      } else {
        const float* fp = tapef + col * DD + koff;
        v16bf bv;
        for (int i = 0; i < 16; i += 4) {
          v4f f = *(const v4f*)(fp + i);
          bv[i] = f2bf(f[0]); bv[i + 1] = f2bf(f[1]);
          bv[i + 2] = f2bf(f[2]); bv[i + 3] = f2bf(f[3]);
        }
        b[t] = bv;
      }
    }
    for (int m = 0; m < 4; ++m)
      for (int t = 0; t < 2; ++t)
        acc[m][t] = __builtin_amdgcn_wmma_f32_16x16x32_bf16(
            false, a[m], false, b[t], (short)0, acc[m][t], false, false);
  }

  // spill raw scores to LDS (C layout: lane half -> M+8, reg r -> M row)
  for (int m = 0; m < 4; ++m)
    for (int t = 0; t < 2; ++t) {
      const int colL = wv * 32 + t * 16 + l16;
      for (int r = 0; r < 8; ++r)
        s_sc[m * 16 + hsel * 8 + r][colL] = acc[m][t][r];
    }
  __syncthreads();

  // exact block-local top-16 per row; wave wv owns rows 8wv..8wv+7.
  // Mask applied from bitset: bit `lane` of word j covers column lane+32j.
  for (int rr = 0; rr < 8; ++rr) {
    const int row = wv * 8 + rr;
    const uint32_t* mwp = maskbits + (size_t)row * MWORDS + (cb >> 5);
    const v4u m0 = *(const v4u*)(mwp);
    const v4u m1 = *(const v4u*)(mwp + 4);
    uint32_t mm[8] = { m0[0], m0[1], m0[2], m0[3], m1[0], m1[1], m1[2], m1[3] };
    float v[8]; int c[8];
    for (int j = 0; j < 8; ++j) {
      const int lc = lane + 32 * j;                   // bank-conflict-free
      v[j] = s_sc[row][lc] - (float)((mm[j] >> lane) & 1u) * 1.0e9f;
      c[j] = cb + lc;
    }
    for (int it = 0; it < KTOP; ++it) {
      float bv = NEG_SENTINEL; int bc = 0x7FFFFFFF;
      for (int j = 0; j < 8; ++j)
        if (v[j] > bv || (v[j] == bv && c[j] < bc)) { bv = v[j]; bc = c[j]; }
      for (int off = 16; off >= 1; off >>= 1) {       // wave32 butterfly argmax
        const float ov = __shfl_xor(bv, off, 32);
        const int   oc = __shfl_xor(bc, off, 32);
        if (ov > bv || (ov == bv && oc < bc)) { bv = ov; bc = oc; }
      }
      if (lane == 0) {
        const int o = (blockIdx.x * BS + row) * KTOP + it;
        cval[o] = bv; cidx[o] = bc;
      }
      for (int j = 0; j < 8; ++j)
        if (c[j] == bc) v[j] = NEG_SENTINEL;          // remove winner
    }
  }
}

// ---------------- per-row: global top-16, softmax, ACT state, tok, q ----
__global__ __launch_bounds__(256) void finish_kernel(
    const float* __restrict__ cval, const int* __restrict__ cidx,
    const float* __restrict__ tapef, const int* __restrict__ flag,
    float* __restrict__ q, __bf16* __restrict__ qb,
    float* __restrict__ hp, float* __restrict__ rem, float* __restrict__ nup,
    uint32_t* __restrict__ maskbits, float* __restrict__ out_toks, int step) {
  if (*flag) return;
  const int r = blockIdx.x;
  const int tid = threadIdx.x;

  __shared__ float cv[NBLK * KTOP];                   // 4096 candidates
  __shared__ int   ci[NBLK * KTOP];
  __shared__ float rv[256]; __shared__ int ri[256];
  __shared__ float s_tval[KTOP]; __shared__ int s_tidx[KTOP];
  __shared__ float s_w[KTOP];

  for (int i = tid; i < NBLK * KTOP; i += 256) {
    const int blk = i >> 4, k = i & 15;
    const int src = (blk * BS + r) * KTOP + k;
    cv[i] = cval[src]; ci[i] = cidx[src];
  }
  __syncthreads();

  for (int it = 0; it < KTOP; ++it) {
    float bv = NEG_SENTINEL; int bc = 0x7FFFFFFF;
    const int base = tid * 16;
    for (int i = base; i < base + 16; ++i)
      if (cv[i] > bv || (cv[i] == bv && ci[i] < bc)) { bv = cv[i]; bc = ci[i]; }
    rv[tid] = bv; ri[tid] = bc;
    __syncthreads();
    for (int s = 128; s >= 1; s >>= 1) {
      if (tid < s &&
          (rv[tid + s] > rv[tid] || (rv[tid + s] == rv[tid] && ri[tid + s] < ri[tid]))) {
        rv[tid] = rv[tid + s]; ri[tid] = ri[tid + s];
      }
      __syncthreads();
    }
    if (tid == 0) { s_tval[it] = rv[0]; s_tidx[it] = ri[0]; }
    __syncthreads();
    const int win = s_tidx[it];
    for (int i = base; i < base + 16; ++i)
      if (ci[i] == win) cv[i] = NEG_SENTINEL;
    __syncthreads();
  }

  if (tid == 0) {
    const float mx = s_tval[0];                        // descending order
    float e[KTOP], sum = 0.0f;
    for (int k = 0; k < KTOP; ++k) { e[k] = __expf((s_tval[k] - mx) * INV_SQRT_D); sum += e[k]; }
    const float inv = 1.0f / sum;
    float ent = 1.0f;
    for (int k = 0; k < KTOP; ++k) { const float w = e[k] * inv; s_w[k] = w; ent -= w * w; }
    // ACT halting algebra (NTPS=1 -> sw = w[0])
    float h = hp[r];
    const float still = (h < 1.0f) ? 1.0f : 0.0f;
    const float sw = s_w[0];
    const float nh = ((h + sw >= 1.0f) ? 1.0f : 0.0f) * still;
    const float st2 = still - nh;
    rem[r] += still * ent;
    h = h + sw * st2;
    h = h + nh * (1.0f - h);
    hp[r] = h;
    nup[r] += still;
  }
  __syncthreads();

  if (tid < KTOP) {                                    // scatter mask bits
    const int idx = s_tidx[tid];
    atomicOr(&maskbits[(size_t)r * MWORDS + (idx >> 5)], 1u << (idx & 31));
  }

  for (int j = tid; j < DD; j += 256) {                // tok gather + q update
    float acc = 0.0f;
    for (int k = 0; k < KTOP; ++k)
      acc += s_w[k] * tapef[(size_t)s_tidx[k] * DD + j];
    out_toks[((size_t)r * STEPS + step) * DD + j] = acc;
    const float nq = (q[r * DD + j] + acc) * 0.5f;
    q[r * DD + j] = nq;
    qb[r * DD + j] = f2bf(nq);
  }
}

// ---------------- all-halted flag (global early-exit) ----------------
__global__ void flag_kernel(const float* __restrict__ hp, int* __restrict__ flag) {
  __shared__ int ok;
  if (threadIdx.x == 0) ok = 1;
  __syncthreads();
  if (threadIdx.x < BS && hp[threadIdx.x] < 1.0f) ok = 0;
  __syncthreads();
  if (threadIdx.x == 0) *flag = ok;
}

// ---------------- final rem/nup write-out ----------------
__global__ void writeout_kernel(const float* __restrict__ rem,
                                const float* __restrict__ nup,
                                float* __restrict__ out) {
  const int i = threadIdx.x;
  if (i < BS) {
    out[(size_t)BS * STEPS * DD + i]      = rem[i];
    out[(size_t)BS * STEPS * DD + BS + i] = nup[i];
  }
}

// ---------------- host launcher ----------------
extern "C" void kernel_launch(void* const* d_in, const int* in_sizes, int n_in,
                              void* d_out, int out_size, void* d_ws, size_t ws_size,
                              hipStream_t stream) {
  const float* q_in = (const float*)d_in[0];
  const float* tape = (const float*)d_in[1];
  float* out = (float*)d_out;

  char* p = (char*)d_ws;
  auto alloc = [&](size_t bytes) -> char* {
    char* r = p; p += (bytes + 255) & ~(size_t)255; return r;
  };
  float*    q     = (float*)   alloc((size_t)BS * DD * 4);
  __bf16*   qb    = (__bf16*)  alloc((size_t)BS * DD * 2);
  float*    hp    = (float*)   alloc(BS * 4);
  float*    rem   = (float*)   alloc(BS * 4);
  float*    nup   = (float*)   alloc(BS * 4);
  int*      flag  = (int*)     alloc(16);
  uint32_t* maskb = (uint32_t*)alloc((size_t)BS * MWORDS * 4);     // 512 KB
  float*    cval  = (float*)   alloc((size_t)NBLK * BS * KTOP * 4);
  int*      cidx  = (int*)     alloc((size_t)NBLK * BS * KTOP * 4);
  const size_t used = (size_t)(p - (char*)d_ws);
  const size_t tape_bytes = (size_t)BANK * DD * 2;     // 96 MB bf16 bank
  const bool pre = (ws_size >= used + tape_bytes + 256);
  __bf16* tapeb = pre ? (__bf16*)alloc(tape_bytes) : nullptr;

  init_kernel<<<2048, 256, 0, stream>>>(q_in, q, qb, hp, rem, nup, flag,
                                        maskb, out);
  if (pre) convert_kernel<<<4096, 256, 0, stream>>>(tape, tapeb);

  for (int s = 0; s < STEPS; ++s) {
    if (pre)
      scores_topk_kernel<true><<<NBLK, 256, 0, stream>>>(qb, tapeb, tape, maskb,
                                                         flag, cval, cidx);
    else
      scores_topk_kernel<false><<<NBLK, 256, 0, stream>>>(qb, tapeb, tape, maskb,
                                                          flag, cval, cidx);
    finish_kernel<<<BS, 256, 0, stream>>>(cval, cidx, tape, flag, q, qb,
                                          hp, rem, nup, maskb, out, s);
    flag_kernel<<<1, 64, 0, stream>>>(hp, flag);
  }
  writeout_kernel<<<1, 64, 0, stream>>>(rem, nup, out);
}
